// AttentionMechanism_12249246728342
// MI455X (gfx1250) — matrix operand order repaired and checked by reading, not compiled, not run on hardware
//
#include <hip/hip_runtime.h>

#define BATCH 4
#define SEQ   2048
#define D_DIM 1024
#define HEADS 16
#define HDIM  64
#define M_TOT (BATCH * SEQ)

typedef __attribute__((ext_vector_type(16))) unsigned short v16u;
typedef __attribute__((ext_vector_type(8)))  unsigned short v8u;
typedef __attribute__((ext_vector_type(4)))  unsigned short v4u;
typedef __attribute__((ext_vector_type(16))) __bf16         v16bf;
typedef __attribute__((ext_vector_type(8)))  float          v8f;
typedef __attribute__((ext_vector_type(4)))  unsigned int   u32x4;
typedef __attribute__((ext_vector_type(4)))  int            i32x4;
typedef __attribute__((ext_vector_type(8)))  int            i32x8;

static_assert(sizeof(v16u) == 32 && sizeof(v16bf) == 32, "frag size");

__device__ __forceinline__ unsigned short f2bf(float f) {
  unsigned int u = __float_as_uint(f);
  u += 0x7FFFu + ((u >> 16) & 1u);   // round-to-nearest-even
  return (unsigned short)(u >> 16);
}

__device__ __forceinline__ v8f wmma_bf16(v16u a, v16u b, v8f c) {
  return __builtin_amdgcn_wmma_f32_16x16x32_bf16(
      false, __builtin_bit_cast(v16bf, a),
      false, __builtin_bit_cast(v16bf, b),
      (short)0, c, false, false);
}

// --- CDNA5 async global->LDS copy (ASYNCcnt), ISA 10.x / 15.18.3 ----------
__device__ __forceinline__ void async_ld_b128(unsigned ldsAddr, const void* g) {
  asm volatile("global_load_async_to_lds_b128 %0, %1, off"
               :: "v"(ldsAddr), "v"(g) : "memory");
}
__device__ __forceinline__ void wait_async0() {
  asm volatile("s_wait_asynccnt 0x0" ::: "memory");
}

// --- CDNA5 Tensor Data Mover: 1-D contiguous tile load (TENSORcnt) --------
// D# per CDNA5 ISA: group0 {count=1, lds_addr,
// global_addr, type=2}; group1 {data_size=2B, tensor_dim0=tile_dim0=nElems,
// tensor_dim1=tile_dim1=1, stride0=nElems}.
#if __has_builtin(__builtin_amdgcn_tensor_load_to_lds)
#define HAVE_TDM 1
#endif

#if defined(HAVE_TDM)
__device__ __forceinline__ void tdm_load_1d(unsigned ldsAddr, const void* g,
                                            unsigned nElemsU16) {
  const unsigned long long ga = (unsigned long long)(uintptr_t)g;
  u32x4 g0;
  g0.x = 1u;                                               // count=1 (valid)
  g0.y = ldsAddr;                                          // lds_addr
  g0.z = (unsigned)ga;                                     // global_addr[31:0]
  g0.w = (unsigned)((ga >> 32) & 0x01FFFFFFu) | (2u << 30);// ga[56:32] | type=2
  i32x8 g1;
  g1[0] = 0x00010000;                   // data_size=1 (2 bytes), no multicast
  g1[1] = (int)(nElemsU16 << 16);       // tensor_dim0[15:0] @ bits 63:48
  g1[2] = 0x00010000;                   // tensor_dim1 = 1   @ bits 95:80
  g1[3] = (int)(nElemsU16 << 16);       // tile_dim0         @ bits 127:112
  g1[4] = 1;                            // tile_dim1 = 1
  g1[5] = (int)nElemsU16;               // tensor_dim0_stride
  g1[6] = 0;
  g1[7] = 0;
  const i32x4 z4 = {0, 0, 0, 0};
#if __has_include(<hip/amd_detail/amd_gfx1250_TDM.h>)
  const i32x8 z8 = {0, 0, 0, 0, 0, 0, 0, 0};
  __builtin_amdgcn_tensor_load_to_lds(g0, g1, z4, z4, z8, 0);   // clang-23 form
#else
  __builtin_amdgcn_tensor_load_to_lds(g0, g1, z4, z4, 0);       // ROCm 7.2 form
#endif
}
#endif

// ---------------------------------------------------------------------------
// GEMM: out[m,n] = sum_k A[m,k] * W[n,k]   (torch Linear: X @ W^T)
//   A_BF16=false : A is fp32, converted to bf16 while staging to LDS
//   A_BF16=true  : A is bf16 -> staged with async global->LDS (no VGPR pass)
//   OUT_SPLIT=true : write bf16 to [B,H,S,Dh] layout, scaled (Q/K/V proj)
//   OUT_SPLIT=false: write fp32 [M,D] with bias (final projection)
// Block: 256 threads (8 waves). Tile 128x128, K step 32. Each wave: 16 rows.
// ---------------------------------------------------------------------------
template <bool A_BF16, bool OUT_SPLIT>
__global__ void __launch_bounds__(256)
gemm_wmma_kernel(const void* __restrict__ Aptr, const float* __restrict__ W,
                 const float* __restrict__ bias, void* __restrict__ Out,
                 float scale) {
  __shared__ unsigned short As[128][40];  // +8 pad, rows 16B aligned (80B)
  __shared__ unsigned short Bs[128][40];  // Bs[n][k] = bf16(W[nBase+n][k0+k])

  const int tid  = threadIdx.x;
  const int lane = tid & 31;
  const int wave = tid >> 5;      // 0..7
  const int ln16 = lane & 15;
  const int hi   = lane >> 4;     // 0/1 (lane half)
  const int mBase = blockIdx.y * 128;
  const int nBase = blockIdx.x * 128;

  const float*          Af = (const float*)Aptr;
  const unsigned short* Ab = (const unsigned short*)Aptr;

  v8f acc[8] = {};

  for (int k0 = 0; k0 < D_DIM; k0 += 32) {
    __syncthreads();
    // --- stage A tile (128x32 bf16) ---
    if (A_BF16) {
      // pure copy: async DMA into LDS, 16B per issue, 2 issues/thread
      for (int t = tid; t < 128 * 4; t += 256) {
        const int r = t >> 2, c = (t & 3) << 3;
        async_ld_b128((unsigned)(uintptr_t)&As[r][c],
                      &Ab[(size_t)(mBase + r) * D_DIM + k0 + c]);
      }
      wait_async0();
    } else {
      for (int t = tid; t < 128 * 8; t += 256) {
        const int r = t >> 3, c = (t & 7) << 2;
        const float4 v =
            *(const float4*)&Af[(size_t)(mBase + r) * D_DIM + k0 + c];
        v4u p; p.x = f2bf(v.x); p.y = f2bf(v.y); p.z = f2bf(v.z); p.w = f2bf(v.w);
        *(v4u*)&As[r][c] = p;
      }
    }
    // --- stage B tile (128 output cols x 32 k) from W rows, fp32->bf16 ---
    for (int t = tid; t < 128 * 8; t += 256) {
      const int r = t >> 3, c = (t & 7) << 2;
      const float4 v = *(const float4*)&W[(size_t)(nBase + r) * D_DIM + k0 + c];
      v4u p; p.x = f2bf(v.x); p.y = f2bf(v.y); p.z = f2bf(v.z); p.w = f2bf(v.w);
      *(v4u*)&Bs[r][c] = p;
    }
    __syncthreads();

    // A fragment: 16x32, row = wave*16 + ln16, K = i + 16*hi
    v16u a;
    {
      const unsigned short* ar = &As[wave * 16 + ln16][0];
#pragma unroll
      for (int i = 0; i < 16; ++i)
        a[i] = ar[i + (hi << 4)];
    }
#pragma unroll
    for (int t = 0; t < 8; ++t) {
      v16u b;
      const unsigned short* br = &Bs[t * 16 + ln16][0];
#pragma unroll
      for (int i = 0; i < 16; ++i) b[i] = br[i + (hi << 4)];  // K = i + 16*hi
      acc[t] = wmma_bf16(a, b, acc[t]);
    }
  }

  // epilogue: C/D layout -> element r is row (r + 8*hi), lane%16 is col
#pragma unroll
  for (int t = 0; t < 8; ++t) {
#pragma unroll
    for (int r = 0; r < 8; ++r) {
      const int gm = mBase + wave * 16 + r + (hi << 3);
      const int gn = nBase + t * 16 + ln16;
      const float v = acc[t][r] * scale;
      if (OUT_SPLIT) {
        const int bb = gm / SEQ, ss = gm % SEQ;
        const int hh = gn >> 6, dh = gn & 63;
        ((unsigned short*)Out)[((size_t)(bb * HEADS + hh) * SEQ + ss) * HDIM + dh] =
            f2bf(v);
      } else {
        ((float*)Out)[(size_t)gm * D_DIM + gn] = v + bias[gn];
      }
    }
  }
}

// ---------------------------------------------------------------------------
// Flash attention: grid (S/64, H, B); 128 threads = 4 waves, 16 q-rows/wave.
// K/V chunks are contiguous 4KB blocks in [B,H,S,Dh] -> staged by the TDM.
// Q was pre-scaled by 1/sqrt(EMBED_DIM) in the projection.
// ---------------------------------------------------------------------------
__global__ void __launch_bounds__(128)
attn_kernel(const unsigned short* __restrict__ Qp,
            const unsigned short* __restrict__ Kp,
            const unsigned short* __restrict__ Vp,
            unsigned short* __restrict__ AttnOut) {
  __shared__ unsigned short Ks[32][64];      // 32 keys x 64 dh (TDM target)
  __shared__ unsigned short Vs[32][64];
  __shared__ unsigned short Ps[4][16][40];   // per-wave P staging (16x32)

  const int tid  = threadIdx.x;
  const int lane = tid & 31;
  const int wave = tid >> 5;   // 0..3
  const int ln16 = lane & 15;
  const int hi   = lane >> 4;
  const int b    = blockIdx.z;
  const int h    = blockIdx.y;
  const int qBase = blockIdx.x * 64 + wave * 16;
  const size_t headOff = (size_t)(b * HEADS + h) * SEQ * HDIM;

  // Q A-fragments: 16 rows x 64 dh -> two 16x32 frags (K steps of 32)
  v16u qa[2];
  {
    const unsigned short* qrow = &Qp[headOff + (size_t)(qBase + ln16) * HDIM];
#pragma unroll
    for (int kk = 0; kk < 2; ++kk)
#pragma unroll
      for (int i = 0; i < 16; ++i)
        qa[kk][i] = qrow[kk * 32 + i + (hi << 4)];
  }

  v8f o[4] = {};
  float mrow[8], lrow[8];
#pragma unroll
  for (int r = 0; r < 8; ++r) { mrow[r] = -3.0e38f; lrow[r] = 0.0f; }

  for (int j = 0; j < SEQ; j += 32) {
    __syncthreads();
#if defined(HAVE_TDM)
    if (wave == 0) {   // TDM issue is per-wave (EXEC ignored)
      tdm_load_1d((unsigned)(uintptr_t)&Ks[0][0],
                  &Kp[headOff + (size_t)j * HDIM], 32 * HDIM);
      tdm_load_1d((unsigned)(uintptr_t)&Vs[0][0],
                  &Vp[headOff + (size_t)j * HDIM], 32 * HDIM);
      __builtin_amdgcn_s_wait_tensorcnt(0);
    }
#else
    for (int t = tid; t < 32 * 8; t += 128) {
      const int r = t >> 3, c = (t & 7) << 3;
      *(v8u*)&Ks[r][c] = *(const v8u*)&Kp[headOff + (size_t)(j + r) * HDIM + c];
      *(v8u*)&Vs[r][c] = *(const v8u*)&Vp[headOff + (size_t)(j + r) * HDIM + c];
    }
#endif
    if (j + 32 < SEQ) {  // hint next chunk (global_prefetch_b8)
      __builtin_prefetch(&Kp[headOff + (size_t)(j + 32 + (tid >> 2)) * HDIM], 0, 1);
      __builtin_prefetch(&Vp[headOff + (size_t)(j + 32 + (tid >> 2)) * HDIM], 0, 1);
    }
    __syncthreads();

    // scores: two 16x16 tiles (keys j..j+15, j+16..j+31), contraction Dh=64
    v8f s0 = {}, s1 = {};
#pragma unroll
    for (int kk = 0; kk < 2; ++kk) {
      v16u b0, b1;
      const unsigned short* k0r = &Ks[ln16][kk * 32 + (hi << 4)];
      const unsigned short* k1r = &Ks[16 + ln16][kk * 32 + (hi << 4)];
#pragma unroll
      for (int i = 0; i < 16; ++i) { b0[i] = k0r[i]; b1[i] = k1r[i]; }
      s0 = wmma_bf16(qa[kk], b0, s0);
      s1 = wmma_bf16(qa[kk], b1, s1);
    }

    // online softmax (rows live across 16-lane halves -> xor-shuffle reduce)
#pragma unroll
    for (int r = 0; r < 8; ++r) {
      float mx = fmaxf(s0[r], s1[r]);
      mx = fmaxf(mx, __shfl_xor(mx, 1, 32));
      mx = fmaxf(mx, __shfl_xor(mx, 2, 32));
      mx = fmaxf(mx, __shfl_xor(mx, 4, 32));
      mx = fmaxf(mx, __shfl_xor(mx, 8, 32));
      const float mnew  = fmaxf(mrow[r], mx);
      const float alpha = __expf(mrow[r] - mnew);
      const float p0 = __expf(s0[r] - mnew);
      const float p1 = __expf(s1[r] - mnew);
      float sum = p0 + p1;
      sum += __shfl_xor(sum, 1, 32);
      sum += __shfl_xor(sum, 2, 32);
      sum += __shfl_xor(sum, 4, 32);
      sum += __shfl_xor(sum, 8, 32);
      lrow[r] = lrow[r] * alpha + sum;
      mrow[r] = mnew;
      s0[r] = p0; s1[r] = p1;
      o[0][r] *= alpha; o[1][r] *= alpha; o[2][r] *= alpha; o[3][r] *= alpha;
    }

    // stage P (C/D layout -> row-major 16x32 bf16 in LDS)
#pragma unroll
    for (int r = 0; r < 8; ++r) {
      Ps[wave][r + (hi << 3)][ln16]      = f2bf(s0[r]);
      Ps[wave][r + (hi << 3)][16 + ln16] = f2bf(s1[r]);
    }

    // reload P as A-fragment, then PV
    v16u pa;
    {
      const unsigned short* pr = &Ps[wave][ln16][0];
#pragma unroll
      for (int i = 0; i < 16; ++i)
        pa[i] = pr[i + (hi << 4)];
    }
#pragma unroll
    for (int t = 0; t < 4; ++t) {
      v16u bv;  // B[k=key][n=dh]: lane n needs Vs[key][t*16+n], K = i + 16*hi
#pragma unroll
      for (int i = 0; i < 16; ++i) bv[i] = Vs[i + (hi << 4)][t * 16 + ln16];
      o[t] = wmma_bf16(pa, bv, o[t]);
    }
  }

  // epilogue: normalize and write [B, S, H*Dh] bf16
#pragma unroll
  for (int t = 0; t < 4; ++t) {
#pragma unroll
    for (int r = 0; r < 8; ++r) {
      const int s    = qBase + r + (hi << 3);
      const int dcol = h * HDIM + t * 16 + ln16;
      AttnOut[(size_t)(b * SEQ + s) * D_DIM + dcol] = f2bf(o[t][r] / lrow[r]);
    }
  }
}

// ---------------------------------------------------------------------------
extern "C" void kernel_launch(void* const* d_in, const int* in_sizes, int n_in,
                              void* d_out, int out_size, void* d_ws, size_t ws_size,
                              hipStream_t stream) {
  (void)in_sizes; (void)n_in; (void)out_size; (void)ws_size;
  const float* values = (const float*)d_in[0];
  const float* keys   = (const float*)d_in[1];
  const float* query  = (const float*)d_in[2];
  const float* Wv     = (const float*)d_in[3];
  const float* Wk     = (const float*)d_in[4];
  const float* Wq     = (const float*)d_in[5];
  const float* Wo     = (const float*)d_in[6];
  const float* bo     = (const float*)d_in[7];

  const size_t elems = (size_t)M_TOT * D_DIM;  // 8M bf16 elems per buffer
  unsigned short* Qp = (unsigned short*)d_ws;
  unsigned short* Kp = Qp + elems;
  unsigned short* Vp = Kp + elems;
  unsigned short* Ao = Vp + elems;             // total 64 MB of d_ws

  const dim3 gProj(D_DIM / 128, M_TOT / 128);  // (8, 64)
  const float qscale = 0.03125f;               // 1/sqrt(1024)

  gemm_wmma_kernel<false, true><<<gProj, 256, 0, stream>>>(query,  Wq, nullptr, Qp, qscale);
  gemm_wmma_kernel<false, true><<<gProj, 256, 0, stream>>>(keys,   Wk, nullptr, Kp, 1.0f);
  gemm_wmma_kernel<false, true><<<gProj, 256, 0, stream>>>(values, Wv, nullptr, Vp, 1.0f);

  const dim3 gAttn(SEQ / 64, HEADS, BATCH);
  attn_kernel<<<gAttn, 128, 0, stream>>>(Qp, Kp, Vp, Ao);

  gemm_wmma_kernel<true, false><<<gProj, 256, 0, stream>>>(Ao, Wo, bo, (float*)d_out, 1.0f);
}
